// DependencyParser_88837103550580
// MI455X (gfx1250) — matrix operand order repaired and checked by reading, not compiled, not run on hardware
//
#include <hip/hip_runtime.h>
#include <hip/hip_bf16.h>

// ---------------------------------------------------------------------------
// DependencyParser (BiLSTM x2 + biaffine scorer) for MI455X / gfx1250.
//   B=16, L=256, D=H=128, G=4H=512.  All hoistable GEMMs use
//   v_wmma_f32_16x16x32_f16 (wave32 WMMA, f32 accumulate).  GEMM K-stages are
//   double-buffered with gfx1250 async global->LDS copies (ASYNCcnt).
//   Sequential recurrence keeps state in LDS; intermediates (~53MB) stay L2-
//   resident (192MB L2), so HBM traffic is only inputs + 4MB output.
// ---------------------------------------------------------------------------

typedef __attribute__((ext_vector_type(16))) _Float16 v16h;
typedef __attribute__((ext_vector_type(8)))  _Float16 v8h;
typedef __attribute__((ext_vector_type(8)))  float    v8f;
typedef __attribute__((ext_vector_type(4)))  int      v4i;

#define BB 16
#define LL 256
#define DD 128
#define HH 128
#define GG 512
#define ML (BB * LL)          // 4096 flattened (b,l) rows

#if defined(__has_builtin)
#if __has_builtin(__builtin_amdgcn_global_load_async_to_lds_b128) && \
    __has_builtin(__builtin_amdgcn_s_wait_asynccnt)
#define USE_ASYNC_LDS 1
#endif
#endif
#ifndef USE_ASYNC_LDS
#define USE_ASYNC_LDS 0
#endif

#define GLOBAL_AS __attribute__((address_space(1)))
#define LDS_AS    __attribute__((address_space(3)))

#if USE_ASYNC_LDS
__device__ __forceinline__ void async_cp_b128(const void* g, void* l) {
  __builtin_amdgcn_global_load_async_to_lds_b128(
      (GLOBAL_AS v4i*)g, (LDS_AS v4i*)l, /*imm_offset=*/0, /*cpol=*/0);
}
#endif

// ---------------- WMMA fragment helpers (wave32 layouts, CDNA5 ISA) --

__device__ __forceinline__ v8f wmma16(v16h a, v16h b, v8f c) {
  return __builtin_amdgcn_wmma_f32_16x16x32_f16(false, a, false, b, (short)0, c, false, false);
}

// A fragment 16x32 f16, row-major source (lda elems). lane = 16*half + m.
// elem j: K = (j<8) ? j+8*half : j+8+8*half  -> two contiguous 8-half chunks.
__device__ __forceinline__ v16h load_a_frag(const _Float16* A, int lda, int lane) {
  const int half_ = (lane >> 4) & 1, m = lane & 15;
  const _Float16* p = A + m * lda + 8 * half_;
  v8h c0 = *(const v8h*)(p);
  v8h c1 = *(const v8h*)(p + 16);
  v16h r;
#pragma unroll
  for (int j = 0; j < 8; ++j) { r[j] = c0[j]; r[8 + j] = c1[j]; }
  return r;
}

// B fragment 32x16 f16 of W^T, taken from row-major W (N x K, ldb elems).
// lane = 16*half + n ; elem j: K = (j<8) ? j+8*half : j+8+8*half (matches A).
__device__ __forceinline__ v16h load_b_frag(const _Float16* W, int ldb, int lane) {
  const int half_ = (lane >> 4) & 1, n = lane & 15;
  const _Float16* p = W + n * ldb + 8 * half_;
  v8h c0 = *(const v8h*)(p);
  v8h c1 = *(const v8h*)(p + 16);
  v16h r;
#pragma unroll
  for (int j = 0; j < 8; ++j) { r[j] = c0[j]; r[8 + j] = c1[j]; }
  return r;
}

__device__ __forceinline__ float sigmoidf_fast(float x) {
  return 1.0f / (1.0f + __expf(-x));
}

// ---------------- prep kernels ----------------------------------------------

__global__ void embed_gather_kernel(const int* __restrict__ widx,
                                    const int* __restrict__ pidx,
                                    const float* __restrict__ wemb,
                                    const float* __restrict__ temb,
                                    _Float16* __restrict__ x16) {
  const int bl = blockIdx.x;            // 0..4095  (b*L + l)
  const int d  = threadIdx.x;           // 0..127
  float v = (d < 100) ? wemb[(size_t)widx[bl] * 100 + d]
                      : temb[(size_t)pidx[bl] * 28 + (d - 100)];
  x16[(size_t)bl * DD + d] = (_Float16)v;
}

__global__ void cvt_f32_to_f16_kernel(const float* __restrict__ src,
                                      _Float16* __restrict__ dst, int n) {
  int i = blockIdx.x * 256 + threadIdx.x;
  if (i < n) dst[i] = (_Float16)src[i];
}

__global__ void bias_sum_kernel(const float* __restrict__ bih,
                                const float* __restrict__ bhh,
                                float* __restrict__ bsum, int n) {
  int i = blockIdx.x * 256 + threadIdx.x;
  if (i < n) bsum[i] = bih[i] + bhh[i];
}

// fc1_w (100 x 512) -> W1pad/W2pad (128 x 256 f16, zero padded rows 100..127)
__global__ void pad_fc1_kernel(const float* __restrict__ fc1_w,
                               _Float16* __restrict__ w1p,
                               _Float16* __restrict__ w2p) {
  int idx = blockIdx.x * 256 + threadIdx.x;   // 128*256
  int r = idx >> 8, k = idx & 255;
  float v1 = (r < 100) ? fc1_w[r * 512 + k] : 0.0f;
  float v2 = (r < 100) ? fc1_w[r * 512 + 256 + k] : 0.0f;
  w1p[idx] = (_Float16)v1;
  w2p[idx] = (_Float16)v2;
}

__global__ void pad_fc1b_kernel(const float* __restrict__ fc1_b,
                                float* __restrict__ bpad) {
  int r = threadIdx.x;                  // 128
  bpad[r] = (r < 100) ? fc1_b[r] : 0.0f;
}

// ---------------- WMMA GEMM:  C(MxN,f32) = A(MxK,f16) * W(NxK,f16)^T + bias -
// Block = 256 threads (8 waves) -> 64x64 output tile.
// K-stages double-buffered: stage s+1 is copied global->LDS asynchronously
// (global_load_async_to_lds_b128 / s_wait_asynccnt) while stage s computes.

#define LDT 40                         // LDS row stride in halfs (pad vs 32)
#define TILE_H (64 * LDT)

__global__ __launch_bounds__(256)
void gemm_wmma_kernel(const _Float16* __restrict__ A,
                      const _Float16* __restrict__ W,
                      const float* __restrict__ bias,   // length N or nullptr
                      float* __restrict__ C,
                      int M, int N, int K) {
  __shared__ _Float16 As[2 * TILE_H];
  __shared__ _Float16 Ws[2 * TILE_H];
  const int tid  = threadIdx.x;
  const int wave = tid >> 5, lane = tid & 31;
  const int m0 = blockIdx.y * 64, n0 = blockIdx.x * 64;
  const int mi  = (wave & 3) * 16;        // wave's M subtile inside block tile
  const int ni0 = (wave >> 2) * 32;       // first of wave's two N subtiles

  v8f acc0 = {}; v8f acc1 = {};
  const int row  = tid >> 2;              // 0..63
  const int col8 = (tid & 3) * 8;         // 0,8,16,24
  const _Float16* gA = A + (size_t)(m0 + row) * K + col8;  // + kb per stage
  const _Float16* gW = W + (size_t)(n0 + row) * K + col8;
  const int lofs = row * LDT + col8;      // this thread's LDS slot

#if USE_ASYNC_LDS
  // prologue: async-copy stage 0 into buffer 0
  async_cp_b128(gA, As + lofs);
  async_cp_b128(gW, Ws + lofs);
  int buf = 0;
  for (int kb = 0; kb < K; kb += 32, buf ^= 1) {
    const int nkb = kb + 32;
    if (nkb < K) {                        // prefetch next stage into other buf
      async_cp_b128(gA + nkb, As + (buf ^ 1) * TILE_H + lofs);
      async_cp_b128(gW + nkb, Ws + (buf ^ 1) * TILE_H + lofs);
      __builtin_amdgcn_s_wait_asynccnt(2);   // current stage's 2 ops done
    } else {
      __builtin_amdgcn_s_wait_asynccnt(0);
    }
    __syncthreads();                      // everyone's stage-kb copies visible
    v16h af  = load_a_frag(As + buf * TILE_H + mi * LDT, LDT, lane);
    v16h bf0 = load_b_frag(Ws + buf * TILE_H + ni0 * LDT, LDT, lane);
    v16h bf1 = load_b_frag(Ws + buf * TILE_H + (ni0 + 16) * LDT, LDT, lane);
    acc0 = wmma16(af, bf0, acc0);
    acc1 = wmma16(af, bf1, acc1);
    __syncthreads();                      // done reading buf before its reuse
  }
#else
  // fallback: register double-buffer (next stage loads overlap WMMAs)
  v8h ra = *(const v8h*)(gA);
  v8h rw = *(const v8h*)(gW);
  for (int kb = 0; kb < K; kb += 32) {
    *(v8h*)(As + lofs) = ra;
    *(v8h*)(Ws + lofs) = rw;
    __syncthreads();
    const int nkb = kb + 32;
    if (nkb < K) { ra = *(const v8h*)(gA + nkb); rw = *(const v8h*)(gW + nkb); }
    v16h af  = load_a_frag(As + mi * LDT, LDT, lane);
    v16h bf0 = load_b_frag(Ws + ni0 * LDT, LDT, lane);
    v16h bf1 = load_b_frag(Ws + (ni0 + 16) * LDT, LDT, lane);
    acc0 = wmma16(af, bf0, acc0);
    acc1 = wmma16(af, bf1, acc1);
    __syncthreads();
  }
#endif

  // D layout: lane = 16*half + n ; vgpr r -> row = r + 8*half
  const int half_ = (lane >> 4) & 1, n = lane & 15;
  const float b0 = bias ? bias[n0 + ni0 + n]      : 0.0f;
  const float b1 = bias ? bias[n0 + ni0 + 16 + n] : 0.0f;
#pragma unroll
  for (int r = 0; r < 8; ++r) {
    const size_t rr = (size_t)(m0 + mi + r + 8 * half_) * N;
    C[rr + n0 + ni0 + n]      = acc0[r] + b0;
    C[rr + n0 + ni0 + 16 + n] = acc1[r] + b1;
  }
}

// ---------------- sequential BiLSTM recurrence ------------------------------
// grid.x = 2 (dir), 256 threads.  Per step: z(16x512) = zin[t] + h16 @ Whh^T
// via 32 WMMA tiles (8 waves x 4 n-tiles x 4 k-steps), then gates on VALU.
// Whh B-fragments stream from global (L2-resident, 128KB/dir).

__global__ __launch_bounds__(256)
void lstm_layer_kernel(const float* __restrict__ zin_base,   // [2][B*L][512]
                       const _Float16* __restrict__ whh_base,// [2][512][128]
                       float* __restrict__ hout,             // [B][L][256]
                       int hout_stride) {
  __shared__ _Float16 h16[BB * HH];     // 4 KB
  __shared__ float    cst[BB * HH];     // 8 KB
  __shared__ float    zsm[BB * GG];     // 32 KB

  const int dir = blockIdx.x;
  const float*    zin = zin_base + (size_t)dir * ML * GG;
  const _Float16* Whh = whh_base + (size_t)dir * GG * HH;
  const int coloff  = dir * HH;
  const int reverse = dir;

  const int tid  = threadIdx.x;
  const int wave = tid >> 5, lane = tid & 31;

  for (int e = tid; e < BB * HH; e += 256) { h16[e] = (_Float16)0.0f; cst[e] = 0.0f; }
  __syncthreads();

  for (int s = 0; s < LL; ++s) {
    const int t = reverse ? (LL - 1 - s) : s;

    v8f acc[4] = {v8f{}, v8f{}, v8f{}, v8f{}};
#pragma unroll
    for (int kb = 0; kb < HH; kb += 32) {
      v16h af = load_a_frag(h16 + kb, HH, lane);      // M rows = batch 0..15
#pragma unroll
      for (int u = 0; u < 4; ++u) {
        const int nb = wave * 64 + u * 16;
        v16h bf = load_b_frag(Whh + (size_t)nb * HH + kb, HH, lane);
        acc[u] = wmma16(af, bf, acc[u]);
      }
    }

    const int half_ = (lane >> 4) & 1, n = lane & 15;
#pragma unroll
    for (int u = 0; u < 4; ++u) {
      const int nb = wave * 64 + u * 16;
#pragma unroll
      for (int r = 0; r < 8; ++r) {
        const int b = r + 8 * half_;
        zsm[b * GG + nb + n] =
            acc[u][r] + zin[((size_t)b * LL + t) * GG + nb + n];
      }
    }
    __syncthreads();

    for (int e = tid; e < BB * HH; e += 256) {
      const int b = e >> 7, hh = e & 127;
      const float* zr = zsm + b * GG;
      float iv = sigmoidf_fast(zr[hh]);
      float fv = sigmoidf_fast(zr[HH + hh]);
      float gv = tanhf(zr[2 * HH + hh]);
      float ov = sigmoidf_fast(zr[3 * HH + hh]);
      float cv = fv * cst[e] + iv * gv;
      float hv = ov * tanhf(cv);
      cst[e] = cv;
      h16[e] = (_Float16)hv;
      hout[((size_t)b * LL + t) * hout_stride + coloff + hh] = hv;
    }
    __syncthreads();
  }
}

// ---------------- biaffine pairwise scorer ----------------------------------
// out[i][b][j] = fc2_b + sum_k<100 fc2_w[k] * tanh(a[b,i,k] + c[b,j,k])

__global__ __launch_bounds__(256)
void pairwise_kernel(const float* __restrict__ a,     // [4096][128]
                     const float* __restrict__ c,     // [4096][128]
                     const float* __restrict__ fc2_w, // [100]
                     const float* __restrict__ fc2_b, // [1]
                     float* __restrict__ out) {
  __shared__ float as_[100];
  __shared__ float ws_[100];
  const int b = blockIdx.x >> 8;          // 0..15
  const int i = blockIdx.x & 255;         // 0..255
  if (threadIdx.x < 100) {
    as_[threadIdx.x] = a[((size_t)b * LL + i) * 128 + threadIdx.x];
    ws_[threadIdx.x] = fc2_w[threadIdx.x];
  }
  __syncthreads();
  const int j = threadIdx.x;
  const float* crow = c + ((size_t)b * LL + j) * 128;
  float s = 0.0f;
#pragma unroll 4
  for (int k = 0; k < 100; ++k) s += ws_[k] * tanhf(as_[k] + crow[k]);
  out[((size_t)i * BB + b) * LL + j] = s + fc2_b[0];
}

// ---------------- host launch -----------------------------------------------

extern "C" void kernel_launch(void* const* d_in, const int* in_sizes, int n_in,
                              void* d_out, int out_size, void* d_ws, size_t ws_size,
                              hipStream_t stream) {
  const int*   widx   = (const int*)d_in[0];
  const int*   pidx   = (const int*)d_in[1];
  // d_in[2] = max_length (256), d_in[3] = lengths (unused by reference)
  const float* wemb   = (const float*)d_in[4];
  const float* temb   = (const float*)d_in[5];
  const float* wih_l0 = (const float*)d_in[6];
  const float* whh_l0 = (const float*)d_in[7];
  const float* bih_l0 = (const float*)d_in[8];
  const float* bhh_l0 = (const float*)d_in[9];
  const float* wih_l1 = (const float*)d_in[10];
  const float* whh_l1 = (const float*)d_in[11];
  const float* bih_l1 = (const float*)d_in[12];
  const float* bhh_l1 = (const float*)d_in[13];
  const float* fc1_w  = (const float*)d_in[14];
  const float* fc1_b  = (const float*)d_in[15];
  const float* fc2_w  = (const float*)d_in[16];
  const float* fc2_b  = (const float*)d_in[17];
  float* out = (float*)d_out;

  // ---- carve workspace (~53 MB) ----
  char* ws = (char*)d_ws;
  size_t off = 0;
  auto carve = [&](size_t bytes) -> char* {
    char* p = ws + off;
    off = (off + bytes + 255) & ~(size_t)255;
    return p;
  };
  _Float16* x16     = (_Float16*)carve((size_t)ML * DD * 2);       // 1 MB
  _Float16* wih0_16 = (_Float16*)carve((size_t)2 * GG * DD * 2);   // 256 KB
  _Float16* whh0_16 = (_Float16*)carve((size_t)2 * GG * HH * 2);   // 256 KB
  _Float16* wih1_16 = (_Float16*)carve((size_t)2 * GG * 256 * 2);  // 512 KB
  _Float16* whh1_16 = (_Float16*)carve((size_t)2 * GG * HH * 2);   // 256 KB
  _Float16* w1p     = (_Float16*)carve((size_t)128 * 256 * 2);     // 64 KB
  _Float16* w2p     = (_Float16*)carve((size_t)128 * 256 * 2);     // 64 KB
  float*    bsum0   = (float*)carve((size_t)2 * GG * 4);
  float*    bsum1   = (float*)carve((size_t)2 * GG * 4);
  float*    fc1bp   = (float*)carve((size_t)128 * 4);
  float*    zin0    = (float*)carve((size_t)2 * ML * GG * 4);      // 16 MB
  float*    zin1    = (float*)carve((size_t)2 * ML * GG * 4);      // 16 MB
  float*    h0cat   = (float*)carve((size_t)ML * 256 * 4);         // 4 MB
  _Float16* h0c16   = (_Float16*)carve((size_t)ML * 256 * 2);      // 2 MB
  float*    h1cat   = (float*)carve((size_t)ML * 256 * 4);         // 4 MB
  _Float16* h1c16   = (_Float16*)carve((size_t)ML * 256 * 2);      // 2 MB
  float*    a_buf   = (float*)carve((size_t)ML * 128 * 4);         // 2 MB
  float*    c_buf   = (float*)carve((size_t)ML * 128 * 4);         // 2 MB
  (void)ws_size; (void)in_sizes; (void)n_in; (void)out_size;

  // ---- prep: gather embeddings, f16 weight conversion, bias sums ----
  embed_gather_kernel<<<ML, 128, 0, stream>>>(widx, pidx, wemb, temb, x16);
  cvt_f32_to_f16_kernel<<<(2*GG*DD + 255)/256, 256, 0, stream>>>(wih_l0, wih0_16, 2*GG*DD);
  cvt_f32_to_f16_kernel<<<(2*GG*HH + 255)/256, 256, 0, stream>>>(whh_l0, whh0_16, 2*GG*HH);
  cvt_f32_to_f16_kernel<<<(2*GG*256 + 255)/256, 256, 0, stream>>>(wih_l1, wih1_16, 2*GG*256);
  cvt_f32_to_f16_kernel<<<(2*GG*HH + 255)/256, 256, 0, stream>>>(whh_l1, whh1_16, 2*GG*HH);
  bias_sum_kernel<<<4, 256, 0, stream>>>(bih_l0, bhh_l0, bsum0, 2*GG);
  bias_sum_kernel<<<4, 256, 0, stream>>>(bih_l1, bhh_l1, bsum1, 2*GG);
  pad_fc1_kernel<<<128, 256, 0, stream>>>(fc1_w, w1p, w2p);
  pad_fc1b_kernel<<<1, 128, 0, stream>>>(fc1_b, fc1bp);

  // ---- layer 0: hoisted input projection zin0 = x @ Wih^T + b  (per dir) ----
  {
    dim3 g(GG/64, ML/64);
    gemm_wmma_kernel<<<g, 256, 0, stream>>>(x16, wih0_16,            bsum0,      zin0,                    ML, GG, DD);
    gemm_wmma_kernel<<<g, 256, 0, stream>>>(x16, wih0_16 + GG*DD,    bsum0 + GG, zin0 + (size_t)ML*GG,    ML, GG, DD);
  }
  lstm_layer_kernel<<<2, 256, 0, stream>>>(zin0, whh0_16, h0cat, 256);
  cvt_f32_to_f16_kernel<<<(ML*256 + 255)/256, 256, 0, stream>>>(h0cat, h0c16, ML*256);

  // ---- layer 1 ----
  {
    dim3 g(GG/64, ML/64);
    gemm_wmma_kernel<<<g, 256, 0, stream>>>(h0c16, wih1_16,           bsum1,      zin1,                   ML, GG, 256);
    gemm_wmma_kernel<<<g, 256, 0, stream>>>(h0c16, wih1_16 + GG*256,  bsum1 + GG, zin1 + (size_t)ML*GG,   ML, GG, 256);
  }
  lstm_layer_kernel<<<2, 256, 0, stream>>>(zin1, whh1_16, h1cat, 256);
  cvt_f32_to_f16_kernel<<<(ML*256 + 255)/256, 256, 0, stream>>>(h1cat, h1c16, ML*256);

  // ---- biaffine projections: a = h@W1^T ; c = h@W2^T + fc1_b (N padded 128) -
  {
    dim3 g(128/64, ML/64);
    gemm_wmma_kernel<<<g, 256, 0, stream>>>(h1c16, w1p, nullptr, a_buf, ML, 128, 256);
    gemm_wmma_kernel<<<g, 256, 0, stream>>>(h1c16, w2p, fc1bp,   c_buf, ML, 128, 256);
  }

  // ---- pairwise tanh reduction (dominant elementwise cost, ~105M tanh) ----
  pairwise_kernel<<<ML, 256, 0, stream>>>(a_buf, c_buf, fc2_w, fc2_b, out);
}
